// Attention_36756330119312
// MI455X (gfx1250) — matrix operand order repaired and checked by reading, not compiled, not run on hardware
//
#include <hip/hip_runtime.h>
#include <hip/hip_bf16.h>

// ---------------- CDNA5 WMMA types ----------------
typedef __attribute__((ext_vector_type(16))) __bf16 v16bf;
typedef __attribute__((ext_vector_type(8)))  float  v8f;

union AF {                 // one 16x32 bf16 A/B fragment (8 VGPRs)
    v16bf        v;
    uint4        q[2];
    unsigned int u[8];
};

#define WMMA_BF16(a, b, c) \
    __builtin_amdgcn_wmma_f32_16x16x32_bf16(false, (a), false, (b), (short)0, (c), false, false)

__device__ __forceinline__ unsigned short f2bf(float f) {
    unsigned u = __float_as_uint(f);
    unsigned r = u + 0x7FFFu + ((u >> 16) & 1u);   // round-to-nearest-even
    return (unsigned short)(r >> 16);
}

// ---------------- constants ----------------
#define B      8
#define N      1024
#define DIM    1024
#define HEADS  16
#define HD     64
#define INNER  (HEADS * HD)        // 1024
#define ROWS   (B * N)             // 8192
#define NQKV   (3 * INNER)         // 3072
#define SCALE  0.125f              // 64^-0.5

// ---------------- 1) fp32 -> bf16 (row-major copy) ----------------
__global__ void cvt_bf16_k(const float* __restrict__ src,
                           unsigned short* __restrict__ dst, int n) {
    for (int i = blockIdx.x * blockDim.x + threadIdx.x; i < n;
         i += gridDim.x * blockDim.x)
        dst[i] = f2bf(src[i]);
}

// ---------------- 2) fp32 [rows][cols] -> bf16 [cols][rows] ----------------
__global__ void transpose_bf16_k(const float* __restrict__ src,
                                 unsigned short* __restrict__ dst,
                                 int rows, int cols) {
    int total = rows * cols;
    for (int i = blockIdx.x * blockDim.x + threadIdx.x; i < total;
         i += gridDim.x * blockDim.x) {
        int r = i / cols, c = i - r * cols;
        dst[c * rows + r] = f2bf(src[i]);
    }
}

// ---------------- 3) QKV GEMM: block = 256 rows x 64 cols -------------------
// B chunk staged in double-buffered LDS (shared by 8 waves); A fragments
// register double-buffered; 8 WMMAs per 32-K chunk per wave.
// Scatter-stores into Q[b,h,n,d], K[b,h,n,d], Vt[b,h,d,n] (bf16).
__global__ __launch_bounds__(256) void gemm_qkv_k(
        const unsigned short* __restrict__ X,   // [ROWS][DIM] bf16
        const unsigned short* __restrict__ Wt,  // [NQKV][DIM] bf16
        unsigned short* __restrict__ Qb,
        unsigned short* __restrict__ Kb,
        unsigned short* __restrict__ Vt) {
    __shared__ unsigned short Blds[2][64 * 32];   // 2 x 4KB

    const int lane = threadIdx.x & 31;
    const int w    = threadIdx.x >> 5;
    const int hi   = lane >> 4;
    const int ln   = lane & 15;
    const int tn   = blockIdx.y * 64;
    const int row0 = blockIdx.x * 256 + w * 32;

    const unsigned short* pa0 = X + (row0 + ln) * DIM + hi * 8;
    const unsigned short* pa1 = pa0 + 16 * DIM;

    // cooperative B staging: 256 threads x 16B = 4KB chunk
    const int bn = threadIdx.x >> 2;
    const int bo = (threadIdx.x & 3) * 8;
    const unsigned short* bsrc = Wt + (tn + bn) * DIM + bo;

    v8f acc[2][4] = {};
    AF a0X, a1X, a0Y, a1Y;

    auto coopB = [&](int buf, int kc) {
        *(uint4*)(&Blds[buf][bn * 32 + bo]) = *(const uint4*)(bsrc + kc);
    };
    auto ldA2 = [&](AF& f0, AF& f1, int kc) {
        f0.q[0] = *(const uint4*)(pa0 + kc);
        f0.q[1] = *(const uint4*)(pa0 + kc + 16);
        f1.q[0] = *(const uint4*)(pa1 + kc);
        f1.q[1] = *(const uint4*)(pa1 + kc + 16);
    };
    auto step = [&](int buf, const AF& a0, const AF& a1) {
        AF bfr[4];
#pragma unroll
        for (int t = 0; t < 4; ++t) {
            const unsigned short* bp = &Blds[buf][(t * 16 + ln) * 32 + hi * 8];
            bfr[t].q[0] = *(const uint4*)(bp);
            bfr[t].q[1] = *(const uint4*)(bp + 16);
        }
#pragma unroll
        for (int t = 0; t < 4; ++t) {
            acc[0][t] = WMMA_BF16(a0.v, bfr[t].v, acc[0][t]);
            acc[1][t] = WMMA_BF16(a1.v, bfr[t].v, acc[1][t]);
        }
    };

    coopB(0, 0);
    ldA2(a0X, a1X, 0);
    __syncthreads();
    for (int kc = 0; kc < DIM; kc += 64) {
        coopB(1, kc + 32);                 // prefetch chunk kc+32 -> buf1
        ldA2(a0Y, a1Y, kc + 32);
        step(0, a0X, a1X);                 // compute chunk kc from buf0
        __syncthreads();
        if (kc + 64 < DIM) {
            coopB(0, kc + 64);             // prefetch chunk kc+64 -> buf0
            ldA2(a0X, a1X, kc + 64);
        }
        step(1, a0Y, a1Y);                 // compute chunk kc+32 from buf1
        __syncthreads();
    }

    // C layout: VGPR g -> row = g + hi*8, col = ln  (within 16x16 subtile)
#pragma unroll
    for (int set = 0; set < 2; ++set) {
        const int rbase = row0 + set * 16 + hi * 8;
#pragma unroll
        for (int t = 0; t < 4; ++t) {
            const int col = tn + t * 16 + ln;
            const int seg = col >> 10;          // 0=Q 1=K 2=V (uniform/subtile)
            const int cc  = col & 1023;
            const int h   = cc >> 6;
            const int dd  = cc & 63;
#pragma unroll
            for (int g = 0; g < 8; ++g) {
                const int r  = rbase + g;
                const int bb = r >> 10;
                const int nn = r & 1023;
                const unsigned short v = f2bf(acc[set][t][g]);
                if (seg == 0)
                    Qb[((bb * HEADS + h) * N + nn) * HD + dd] = v;
                else if (seg == 1)
                    Kb[((bb * HEADS + h) * N + nn) * HD + dd] = v;
                else
                    Vt[((bb * HEADS + h) * HD + dd) * N + nn] = v;
            }
        }
    }
}

// ---------------- 4) flash attention: 8 waves x 16 rows per block -----------
// K and Vt 32-key tiles double-buffered in LDS, shared by all 8 waves.
__global__ __launch_bounds__(256) void attn_k(
        const unsigned short* __restrict__ Qb,
        const unsigned short* __restrict__ Kb,
        const unsigned short* __restrict__ Vt,
        unsigned short* __restrict__ A2) {      // [ROWS][INNER] bf16
    __shared__ unsigned short Klds[2][32 * 64]; // 2 x 4KB
    __shared__ unsigned short Vlds[2][64 * 32]; // 2 x 4KB
    __shared__ unsigned int   Plds[8][16 * 16]; // per-wave P tile, 8KB

    const int lane = threadIdx.x & 31;
    const int w    = threadIdx.x >> 5;
    const int hi   = lane >> 4;
    const int ln   = lane & 15;
    const int blk  = blockIdx.x;                // b*128 + h*8 + rb
    const int rb   = blk & 7;
    const int h    = (blk >> 3) & 15;
    const int b    = blk >> 7;
    const int rt16 = rb * 128 + w * 16;         // this wave's 16-row tile

    const unsigned short* kbase = Kb + (b * HEADS + h) * N * HD;
    const unsigned short* vbase = Vt + (b * HEADS + h) * HD * N;

    // cooperative staging: one 16B load each for K and V per thread
    const int kk  = threadIdx.x >> 3, kdo = (threadIdx.x & 7) * 8;
    const int vd  = threadIdx.x >> 2, vso = (threadIdx.x & 3) * 8;
    auto stage = [&](int buf, int j0) {
        *(uint4*)(&Klds[buf][kk * 64 + kdo]) =
            *(const uint4*)(kbase + (j0 + kk) * HD + kdo);
        *(uint4*)(&Vlds[buf][vd * 32 + vso]) =
            *(const uint4*)(vbase + vd * N + j0 + vso);
    };

    // Q A-fragments (reused across all key blocks)
    const unsigned short* qrow = Qb + ((b * HEADS + h) * N + rt16 + ln) * HD;
    AF aq[2];
#pragma unroll
    for (int c = 0; c < 2; ++c) {
        const unsigned short* p = qrow + c * 32 + hi * 8;
        aq[c].q[0] = *(const uint4*)(p);
        aq[c].q[1] = *(const uint4*)(p + 16);
    }

    float m[8], l[8];
#pragma unroll
    for (int g = 0; g < 8; ++g) { m[g] = -INFINITY; l[g] = 0.f; }
    v8f acc[4] = {};

    stage(0, 0);
    __syncthreads();

    for (int j0 = 0; j0 < N; j0 += 32) {
        const int cur = (j0 >> 5) & 1;
        if (j0 + 32 < N) stage(cur ^ 1, j0 + 32);   // prefetch next tile

        // ---- K fragments from LDS, then 4 WMMAs for S ----
        AF bk[2][2];
#pragma unroll
        for (int t = 0; t < 2; ++t)
#pragma unroll
            for (int c = 0; c < 2; ++c) {
                const unsigned short* p =
                    &Klds[cur][(t * 16 + ln) * 64 + c * 32 + hi * 8];
                bk[t][c].q[0] = *(const uint4*)(p);
                bk[t][c].q[1] = *(const uint4*)(p + 16);
            }
        v8f s[2] = {};
#pragma unroll
        for (int t = 0; t < 2; ++t)
#pragma unroll
            for (int c = 0; c < 2; ++c)
                s[t] = WMMA_BF16(aq[c].v, bk[t][c].v, s[t]);

        // ---- online softmax (rows pinned per (g,hi); reduce over 16 lanes) ----
        float corr[8];
#pragma unroll
        for (int g = 0; g < 8; ++g) {
            const float a0 = s[0][g] * SCALE;
            const float a1 = s[1][g] * SCALE;
            s[0][g] = a0; s[1][g] = a1;
            float mx = fmaxf(a0, a1);
            mx = fmaxf(mx, __shfl_xor(mx, 1, 32));
            mx = fmaxf(mx, __shfl_xor(mx, 2, 32));
            mx = fmaxf(mx, __shfl_xor(mx, 4, 32));
            mx = fmaxf(mx, __shfl_xor(mx, 8, 32));
            const float mn = fmaxf(m[g], mx);
            corr[g] = __expf(m[g] - mn);
            m[g] = mn;
        }
        unsigned short* ldsS = (unsigned short*)&Plds[w][0];
#pragma unroll
        for (int g = 0; g < 8; ++g) {
            const float p0 = __expf(s[0][g] - m[g]);
            const float p1 = __expf(s[1][g] - m[g]);
            float rs = p0 + p1;
            rs += __shfl_xor(rs, 1, 32);
            rs += __shfl_xor(rs, 2, 32);
            rs += __shfl_xor(rs, 4, 32);
            rs += __shfl_xor(rs, 8, 32);
            l[g] = l[g] * corr[g] + rs;
#pragma unroll
            for (int t = 0; t < 4; ++t) acc[t][g] *= corr[g];
            ldsS[(g + hi * 8) * 32 + ln]      = f2bf(p0);
            ldsS[(g + hi * 8) * 32 + 16 + ln] = f2bf(p1);
        }

        // ---- V fragments from LDS (issued before P readback) ----
        AF bv[4];
#pragma unroll
        for (int td = 0; td < 4; ++td) {
            const unsigned short* p = &Vlds[cur][(td * 16 + ln) * 32 + hi * 8];
            bv[td].q[0] = *(const uint4*)(p);
            bv[td].q[1] = *(const uint4*)(p + 16);
        }
        // ---- relayout P (C-layout) -> A-fragment; DS is in-order per wave ----
        AF pf;
        {
            const unsigned int* rowp = &Plds[w][0] + ln * 16;
            const int h4 = hi * 4;
            pf.u[0] = rowp[0 + h4];  pf.u[1] = rowp[1 + h4];
            pf.u[2] = rowp[2 + h4];  pf.u[3] = rowp[3 + h4];
            pf.u[4] = rowp[8 + h4];  pf.u[5] = rowp[9 + h4];
            pf.u[6] = rowp[10 + h4]; pf.u[7] = rowp[11 + h4];
        }
        // ---- O += P V ----
#pragma unroll
        for (int td = 0; td < 4; ++td)
            acc[td] = WMMA_BF16(pf.v, bv[td].v, acc[td]);

        __syncthreads();   // next-buffer writes done; current-buffer reads done
    }

    // ---- normalize and store to [b, n, h*d] ----
    const int nbase = rt16 + hi * 8;
#pragma unroll
    for (int td = 0; td < 4; ++td) {
        const int col = h * HD + td * 16 + ln;
#pragma unroll
        for (int g = 0; g < 8; ++g) {
            const float v = acc[td][g] / l[g];
            A2[(b * N + nbase + g) * INNER + col] = f2bf(v);
        }
    }
}

// ---------------- 5) output GEMM: A2 @ WoutT^T + bias -> fp32 ----------------
__global__ __launch_bounds__(256) void gemm_out_k(
        const unsigned short* __restrict__ A2,   // [ROWS][INNER] bf16
        const unsigned short* __restrict__ Wt,   // [DIM][INNER]  bf16
        const float* __restrict__ bias,
        float* __restrict__ out) {               // [ROWS][DIM] fp32
    __shared__ unsigned short Blds[2][64 * 32];

    const int lane = threadIdx.x & 31;
    const int w    = threadIdx.x >> 5;
    const int hi   = lane >> 4;
    const int ln   = lane & 15;
    const int tn   = blockIdx.y * 64;
    const int row0 = blockIdx.x * 256 + w * 32;

    const unsigned short* pa0 = A2 + (row0 + ln) * INNER + hi * 8;
    const unsigned short* pa1 = pa0 + 16 * INNER;

    const int bn = threadIdx.x >> 2;
    const int bo = (threadIdx.x & 3) * 8;
    const unsigned short* bsrc = Wt + (tn + bn) * INNER + bo;

    v8f acc[2][4] = {};
    AF a0X, a1X, a0Y, a1Y;

    auto coopB = [&](int buf, int kc) {
        *(uint4*)(&Blds[buf][bn * 32 + bo]) = *(const uint4*)(bsrc + kc);
    };
    auto ldA2 = [&](AF& f0, AF& f1, int kc) {
        f0.q[0] = *(const uint4*)(pa0 + kc);
        f0.q[1] = *(const uint4*)(pa0 + kc + 16);
        f1.q[0] = *(const uint4*)(pa1 + kc);
        f1.q[1] = *(const uint4*)(pa1 + kc + 16);
    };
    auto step = [&](int buf, const AF& a0, const AF& a1) {
        AF bfr[4];
#pragma unroll
        for (int t = 0; t < 4; ++t) {
            const unsigned short* bp = &Blds[buf][(t * 16 + ln) * 32 + hi * 8];
            bfr[t].q[0] = *(const uint4*)(bp);
            bfr[t].q[1] = *(const uint4*)(bp + 16);
        }
#pragma unroll
        for (int t = 0; t < 4; ++t) {
            acc[0][t] = WMMA_BF16(a0.v, bfr[t].v, acc[0][t]);
            acc[1][t] = WMMA_BF16(a1.v, bfr[t].v, acc[1][t]);
        }
    };

    coopB(0, 0);
    ldA2(a0X, a1X, 0);
    __syncthreads();
    for (int kc = 0; kc < INNER; kc += 64) {
        coopB(1, kc + 32);
        ldA2(a0Y, a1Y, kc + 32);
        step(0, a0X, a1X);
        __syncthreads();
        if (kc + 64 < INNER) {
            coopB(0, kc + 64);
            ldA2(a0X, a1X, kc + 64);
        }
        step(1, a0Y, a1Y);
        __syncthreads();
    }

#pragma unroll
    for (int set = 0; set < 2; ++set) {
        const int rbase = row0 + set * 16 + hi * 8;
#pragma unroll
        for (int t = 0; t < 4; ++t) {
            const int col = tn + t * 16 + ln;
            const float bv = bias[col];
#pragma unroll
            for (int g = 0; g < 8; ++g)
                out[(rbase + g) * DIM + col] = acc[set][t][g] + bv;
        }
    }
}

// ---------------- launch ----------------
extern "C" void kernel_launch(void* const* d_in, const int* in_sizes, int n_in,
                              void* d_out, int out_size, void* d_ws, size_t ws_size,
                              hipStream_t stream) {
    const float* x     = (const float*)d_in[0];   // [8,1024,1024]
    const float* w_qkv = (const float*)d_in[1];   // [1024,3072]
    const float* w_out = (const float*)d_in[2];   // [1024,1024]
    const float* b_out = (const float*)d_in[3];   // [1024]
    float* out = (float*)d_out;

    // workspace layout (bf16 stored as u16)
    char* ws = (char*)d_ws;
    unsigned short* Xbf   = (unsigned short*)(ws + 0);           // 16 MB (reused as A2)
    unsigned short* WqkvT = (unsigned short*)(ws + (16u << 20)); //  6 MB
    unsigned short* WoutT = (unsigned short*)(ws + (22u << 20)); //  2 MB
    unsigned short* Qb    = (unsigned short*)(ws + (24u << 20)); // 16 MB
    unsigned short* Kb    = (unsigned short*)(ws + (40u << 20)); // 16 MB
    unsigned short* Vt    = (unsigned short*)(ws + (56u << 20)); // 16 MB
    unsigned short* A2    = Xbf;  // safe reuse: Xbf dead after gemm_qkv_k

    cvt_bf16_k<<<8192, 256, 0, stream>>>(x, Xbf, ROWS * DIM);
    transpose_bf16_k<<<4096, 256, 0, stream>>>(w_qkv, WqkvT, DIM, NQKV);
    transpose_bf16_k<<<2048, 256, 0, stream>>>(w_out, WoutT, INNER, DIM);

    gemm_qkv_k<<<dim3(ROWS / 256, NQKV / 64), 256, 0, stream>>>(Xbf, WqkvT, Qb, Kb, Vt);

    attn_k<<<B * HEADS * (N / 128), 256, 0, stream>>>(Qb, Kb, Vt, A2);

    gemm_out_k<<<dim3(ROWS / 256, DIM / 64), 256, 0, stream>>>(A2, WoutT, b_out, out);
}
